// RNNDecoder_19722489823273
// MI455X (gfx1250) — matrix operand order repaired and checked by reading, not compiled, not run on hardware
//
#include <hip/hip_runtime.h>
#include <hip/hip_bf16.h>

// ---------------------------------------------------------------------------
// Greedy LSTM decoder for MI455X (gfx1250, wave32, WMMA + TDM).
//
// Roofline: B=32 -> 64 FLOP per weight element; weight streaming dominates.
//  * Weights converted ONCE to bf16 (~79MB total, L2-resident for all 100
//    steps) and fed to v_wmma_f32_16x16x32_bf16 with fp32 accumulate.
//  * A-matrix (32xK activations) is DMA'd into LDS once per workgroup by the
//    Tensor Data Mover (tensor_load_to_lds, TENSORcnt), so the vector-memory
//    pipe is dedicated to streaming W panels (global_load_b128 + prefetch).
//  * Per step: embed/concat -> gates WMMA GEMM -> LSTM pointwise ->
//    logits WMMA GEMM (into d_out projected slice) -> fused log_softmax +
//    argmax (wave32 shfl reductions) -> next index.  Sequential dependence is
//    expressed as stream-ordered launches (graph capture amortizes cost).
// ---------------------------------------------------------------------------

static constexpr int kV = 32000;
static constexpr int kE = 512;
static constexpr int kH = 1024;
static constexpr int kI = 1024;
static constexpr int kC = 64;
static constexpr int kB = 32;
static constexpr int kT = 100;
static constexpr int kX = kE + kC;   // 576, concat width

typedef __attribute__((ext_vector_type(16))) unsigned short v16u;
typedef __attribute__((ext_vector_type(16))) __bf16        v16bf;
typedef __attribute__((ext_vector_type(8)))  float         v8f;
typedef __attribute__((ext_vector_type(4)))  unsigned int  v4u;
typedef __attribute__((ext_vector_type(8)))  unsigned int  su8;

union Frag { v16u u; v16bf b; v4u q[2]; };

__device__ __forceinline__ unsigned short f2bf(float f) {
  unsigned int u = __float_as_uint(f);
  u += 0x7fffu + ((u >> 16) & 1u);           // round-to-nearest-even
  return (unsigned short)(u >> 16);
}

__device__ __forceinline__ float sigmoidf_(float x) {
  return 1.f / (1.f + expf(-x));
}

// ---------------------------------------------------------------------------
// Tensor Data Mover: DMA a [rows x K] bf16 row-major tile global -> LDS.
// D# per CDNA5 ISA ch.8:
//   group0: [1:0]=count=1 | [63:32]=lds_addr | [120:64]=global_addr | type=2
//   group1: data_size=1 (2B), tensor_dim0=K, tensor_dim1=rows,
//           tile_dim0=K, tile_dim1=rows, tensor_dim0_stride=K
// Wave-level op (EXEC ignored); completion via TENSORcnt.
// ---------------------------------------------------------------------------
__device__ __forceinline__ void tdm_load_2d(const void* gptr, unsigned lds_byte_off,
                                            unsigned K, unsigned rows) {
  unsigned long long ga = (unsigned long long)(size_t)gptr;
  v4u g0;
  g0[0] = 1u;                                           // count=1, user descriptor
  g0[1] = lds_byte_off;                                 // lds_addr (bytes)
  g0[2] = (unsigned)(ga & 0xffffffffu);                 // global_addr[31:0]
  g0[3] = (unsigned)((ga >> 32) & 0x01ffffffu) | (2u << 30);  // [120:96] | type=2
  su8 g1;
  g1[0] = 0x00010000u;                                  // data_size=1 (2B)
  g1[1] = (K & 0xffffu) << 16;                          // tensor_dim0[15:0]
  g1[2] = ((K >> 16) & 0xffffu) | (rows << 16);         // dim0[31:16] | dim1[15:0]
  g1[3] = K << 16;                                      // dim1[31:16]=0 | tile_dim0=K
  g1[4] = rows;                                         // tile_dim1=rows, tile_dim2=0
  g1[5] = K;                                            // tensor_dim0_stride[31:0]
  g1[6] = 0u;                                           // stride hi / dim1_stride lo
  g1[7] = 0u;
  asm volatile("tensor_load_to_lds %0, %1" :: "s"(g0), "s"(g1) : "memory");
}

// ---------------------------------------------------------------------------
// fp32 -> bf16 bulk conversion (one-time weight repack)
// ---------------------------------------------------------------------------
__global__ __launch_bounds__(256) void cvt_bf16_kernel(
    const float* __restrict__ src, unsigned short* __restrict__ dst, int n) {
  int i = blockIdx.x * 256 + threadIdx.x;
  if (i < n) dst[i] = f2bf(src[i]);
}

// bias = b_ih + b_hh ; idx[b] = BOS(0)
__global__ __launch_bounds__(256) void prep_kernel(
    const float* __restrict__ b_ih, const float* __restrict__ b_hh,
    float* __restrict__ bias, int* __restrict__ idx) {
  int i = blockIdx.x * 256 + threadIdx.x;
  if (i < 4 * kH) bias[i] = b_ih[i] + b_hh[i];
  if (i < kB) idx[i] = 0;
}

// ---------------------------------------------------------------------------
// WMMA GEMM:  out[32, N] = act( X1[32,K1] @ W1[N,K1]^T
//                              (+ X2[32,K2] @ W2[N,K2]^T) + bias[N] )
// bf16 inputs, fp32 accumulate/output; optional bf16 shadow of the output.
// 128 threads = 4 waves; each wave owns a 16-column panel, two accumulators
// cover M rows 0..15 / 16..31.  A comes from LDS (TDM-staged, ds_load),
// W streams from L2 (global_load_b128).  Layouts per CDNA5 ISA 7.12.2.
// ---------------------------------------------------------------------------
__device__ __forceinline__ void mma_panel(
    const unsigned short* __restrict__ Xlds, const unsigned short* __restrict__ W,
    int K, int col0, int lane, v8f& acc0, v8f& acc1) {
  const int hi     = lane >> 4;          // lane group 0 / 1
  const int khalfA = hi << 3;            // A lane-k offset: 0 or 8
  const int kbaseB = hi << 3;            // B lane-k offset: 0 or 8
  const int n      = col0 + (lane & 15);
  const unsigned short* wrow = W + (size_t)n * K + kbaseB;
  const unsigned short* x0   = Xlds + (size_t)(lane & 15) * K + khalfA;
  const unsigned short* x1   = x0 + (size_t)16 * K;

  for (int kb = 0; kb < K; kb += 32) {
    // B fragment: K = kb+kbase+{0..7} and kb+16+kbase+{0..7} -> 2x global_load_b128
    Frag bf; bf.q[0] = *(const v4u*)(wrow + kb);
             bf.q[1] = *(const v4u*)(wrow + kb + 16);
    __builtin_prefetch(wrow + kb + 128, 0, 3);   // W four K-tiles (128 elems) ahead (L2-resident)
    // A fragments from LDS: K = kb+khalf+{0..7} and kb+16+khalf+{0..7}
    Frag a0; a0.q[0] = *(const v4u*)(x0 + kb);
             a0.q[1] = *(const v4u*)(x0 + kb + 16);
    Frag a1; a1.q[0] = *(const v4u*)(x1 + kb);
             a1.q[1] = *(const v4u*)(x1 + kb + 16);
    acc0 = __builtin_amdgcn_wmma_f32_16x16x32_bf16(false, a0.b, false, bf.b,
                                                   (short)0, acc0, false, false);
    acc1 = __builtin_amdgcn_wmma_f32_16x16x32_bf16(false, a1.b, false, bf.b,
                                                   (short)0, acc1, false, false);
  }
}

__global__ __launch_bounds__(128) void wmma_gemm_kernel(
    const unsigned short* __restrict__ X1, const unsigned short* __restrict__ W1, int K1,
    const unsigned short* __restrict__ X2, const unsigned short* __restrict__ W2, int K2,
    const float* __restrict__ bias, float* __restrict__ out,
    unsigned short* __restrict__ out_bf, int N, int act_tanh) {
  extern __shared__ unsigned short smem[];   // [32*K1] (+ [32*K2]) bf16 A panels
  const int lane = threadIdx.x & 31;
  const int wave = threadIdx.x >> 5;
  const int col0 = blockIdx.x * 64 + wave * 16;

  // One wave DMAs the shared A panel(s) into LDS via the Tensor Data Mover,
  // waits on TENSORcnt, then the workgroup barrier publishes LDS to all waves.
  if (wave == 0) {
    const unsigned lds0 = (unsigned)(size_t)&smem[0];
    tdm_load_2d(X1, lds0, (unsigned)K1, 32u);
    if (X2) tdm_load_2d(X2, lds0 + (unsigned)(32 * K1 * 2), (unsigned)K2, 32u);
    __builtin_amdgcn_s_wait_tensorcnt(0);
  }
  __syncthreads();

  v8f acc0 = {}; v8f acc1 = {};
  mma_panel(smem, W1, K1, col0, lane, acc0, acc1);
  if (X2) mma_panel(smem + (size_t)32 * K1, W2, K2, col0, lane, acc0, acc1);

  // C/D layout (32-bit 16x16): lane&15 = column, VGPR r = row r (+8 for hi lanes)
  const int n   = col0 + (lane & 15);
  const int mr0 = (lane >> 4) << 3;      // 0 or 8
  const float bv = bias[n];
#pragma unroll
  for (int r = 0; r < 8; ++r) {
    float v0 = acc0[r] + bv;
    float v1 = acc1[r] + bv;
    if (act_tanh) { v0 = tanhf(v0); v1 = tanhf(v1); }
    const size_t o0 = (size_t)(mr0 + r) * N + n;
    const size_t o1 = (size_t)(16 + mr0 + r) * N + n;
    out[o0] = v0;
    out[o1] = v1;
    if (out_bf) { out_bf[o0] = f2bf(v0); out_bf[o1] = f2bf(v1); }
  }
}

// ---------------------------------------------------------------------------
// x_bf[b, 0:512] = bf16(emb_table[idx[b]])  ;  x_bf[b, 512:576] = bf16(to_concat[b])
// ---------------------------------------------------------------------------
__global__ __launch_bounds__(256) void embed_concat_kernel(
    const float* __restrict__ emb, const float* __restrict__ toc,
    const int* __restrict__ idx, unsigned short* __restrict__ x_bf) {
  int t = blockIdx.x * 256 + threadIdx.x;
  if (t >= kB * kX) return;
  int b = t / kX, j = t - b * kX;
  float v = (j < kE) ? emb[(size_t)idx[b] * kE + j] : toc[b * kC + (j - kE)];
  x_bf[t] = f2bf(v);
}

// ---------------------------------------------------------------------------
// LSTM cell pointwise (gate order i,f,g,o): updates c, emits h (fp32 to
// d_out outputs[t], bf16 shadow for next step's GEMMs).
// ---------------------------------------------------------------------------
__global__ __launch_bounds__(256) void lstm_cell_kernel(
    const float* __restrict__ gates, float* __restrict__ c,
    unsigned short* __restrict__ h_bf, float* __restrict__ h_out) {
  int t = blockIdx.x * 256 + threadIdx.x;
  if (t >= kB * kH) return;
  int b = t >> 10;                // H = 1024
  int j = t & (kH - 1);
  const float* g = gates + (size_t)b * 4 * kH;
  float ig = sigmoidf_(g[j]);
  float fg = sigmoidf_(g[j + kH]);
  float gg = tanhf(g[j + 2 * kH]);
  float og = sigmoidf_(g[j + 3 * kH]);
  float cn = fg * c[t] + ig * gg;
  float hn = og * tanhf(cn);
  c[t]    = cn;
  h_bf[t] = f2bf(hn);
  h_out[t] = hn;
}

// ---------------------------------------------------------------------------
// In-place log_softmax over a [B, V] slice of d_out + argmax -> idx.
// One block per batch row; wave32 shfl_xor reductions + tiny LDS combine.
// ---------------------------------------------------------------------------
__global__ __launch_bounds__(256) void logsoftmax_argmax_kernel(
    float* __restrict__ logits, int* __restrict__ idx) {
  const int b = blockIdx.x;
  float* row = logits + (size_t)b * kV;
  const int tid = threadIdx.x, lane = tid & 31, wid = tid >> 5;
  __shared__ float s_v[8];
  __shared__ int   s_i[8];
  __shared__ float s_max, s_logsum;
  __shared__ int   s_arg;

  // pass 1: max + first-argmax
  float vmax = -3.402823466e38f; int imax = 0x7fffffff;
  for (int j = tid; j < kV; j += 256) {
    float v = row[j];
    if (v > vmax) { vmax = v; imax = j; }
  }
  for (int o = 16; o > 0; o >>= 1) {
    float ov = __shfl_xor(vmax, o, 32);
    int   oi = __shfl_xor(imax, o, 32);
    if (ov > vmax || (ov == vmax && oi < imax)) { vmax = ov; imax = oi; }
  }
  if (lane == 0) { s_v[wid] = vmax; s_i[wid] = imax; }
  __syncthreads();
  if (tid == 0) {
    float bv = s_v[0]; int bi = s_i[0];
    for (int w = 1; w < 8; ++w)
      if (s_v[w] > bv || (s_v[w] == bv && s_i[w] < bi)) { bv = s_v[w]; bi = s_i[w]; }
    s_max = bv; s_arg = bi;
  }
  __syncthreads();
  const float m = s_max;

  // pass 2: sum(exp(x - m))
  float sum = 0.f;
  for (int j = tid; j < kV; j += 256) sum += expf(row[j] - m);
  for (int o = 16; o > 0; o >>= 1) sum += __shfl_xor(sum, o, 32);
  if (lane == 0) s_v[wid] = sum;
  __syncthreads();
  if (tid == 0) {
    float s = 0.f;
    for (int w = 0; w < 8; ++w) s += s_v[w];
    s_logsum = logf(s);
  }
  __syncthreads();
  const float ls = s_logsum;

  // pass 3: write logp in place
  for (int j = tid; j < kV; j += 256) row[j] = row[j] - m - ls;
  if (tid == 0) idx[b] = s_arg;
}

// ---------------------------------------------------------------------------
extern "C" void kernel_launch(void* const* d_in, const int* in_sizes, int n_in,
                              void* d_out, int out_size, void* d_ws, size_t ws_size,
                              hipStream_t stream) {
  (void)in_sizes; (void)n_in; (void)out_size; (void)ws_size;

  const float* init_hidden = (const float*)d_in[1];
  const float* to_concat   = (const float*)d_in[2];
  const float* emb_table   = (const float*)d_in[4];
  const float* bridge_W    = (const float*)d_in[5];
  const float* bridge_b    = (const float*)d_in[6];
  const float* W_ih        = (const float*)d_in[7];
  const float* W_hh        = (const float*)d_in[8];
  const float* b_ih        = (const float*)d_in[9];
  const float* b_hh        = (const float*)d_in[10];
  const float* out_W       = (const float*)d_in[11];
  const float* out_b       = (const float*)d_in[12];
  float* out = (float*)d_out;

  // d_out layout (flat, return order): outputs[T,B,H] | hT[B,H] | cT[B,H] | projected[T,B,V]
  const size_t OUT_HT   = (size_t)kT * kB * kH;
  const size_t OUT_CT   = OUT_HT + (size_t)kB * kH;
  const size_t OUT_PROJ = OUT_CT + (size_t)kB * kH;

  // workspace carve-out (256B aligned)
  char* w = (char*)d_ws;
  auto carve = [&](size_t bytes) { char* p = w; w += (bytes + 255) & ~(size_t)255; return p; };
  unsigned short* outW_bf = (unsigned short*)carve((size_t)kV * kH * 2);   // 65.5 MB
  unsigned short* Wih_bf  = (unsigned short*)carve((size_t)4 * kH * kX * 2);
  unsigned short* Whh_bf  = (unsigned short*)carve((size_t)4 * kH * kH * 2);
  unsigned short* brW_bf  = (unsigned short*)carve((size_t)kH * kI * 2);
  unsigned short* ih_bf   = (unsigned short*)carve((size_t)kB * kI * 2);
  unsigned short* x_bf    = (unsigned short*)carve((size_t)kB * kX * 2);
  unsigned short* h_bf    = (unsigned short*)carve((size_t)kB * kH * 2);
  float*          c_ws    = (float*)carve((size_t)kB * kH * 4);
  float*          gates   = (float*)carve((size_t)kB * 4 * kH * 4);
  float*          bias    = (float*)carve((size_t)4 * kH * 4);
  int*            idx     = (int*)carve((size_t)kB * 4);

  auto blk = [](size_t n) { return (unsigned)((n + 255) / 256); };

  // dynamic LDS sizes for the TDM-staged A panels
  const unsigned lds_bridge = (unsigned)(32 * kI * 2);          // 64 KB
  const unsigned lds_gates  = (unsigned)(32 * (kX + kH) * 2);   // 100 KB
  const unsigned lds_logits = (unsigned)(32 * kH * 2);          // 64 KB

  // --- one-time: repack weights to bf16 (L2-resident thereafter) ---
  cvt_bf16_kernel<<<blk((size_t)kV * kH), 256, 0, stream>>>(out_W, outW_bf, kV * kH);
  cvt_bf16_kernel<<<blk((size_t)4 * kH * kX), 256, 0, stream>>>(W_ih, Wih_bf, 4 * kH * kX);
  cvt_bf16_kernel<<<blk((size_t)4 * kH * kH), 256, 0, stream>>>(W_hh, Whh_bf, 4 * kH * kH);
  cvt_bf16_kernel<<<blk((size_t)kH * kI), 256, 0, stream>>>(bridge_W, brW_bf, kH * kI);
  cvt_bf16_kernel<<<blk((size_t)kB * kI), 256, 0, stream>>>(init_hidden, ih_bf, kB * kI);
  prep_kernel<<<16, 256, 0, stream>>>(b_ih, b_hh, bias, idx);

  // --- bridge: h0 = c0 = tanh(init_hidden @ bridge_W^T + bridge_b) ---
  // fp32 result -> c_ws (c0); bf16 shadow -> h_bf (feeds first gates GEMM)
  wmma_gemm_kernel<<<kH / 64, 128, lds_bridge, stream>>>(
      ih_bf, brW_bf, kI, nullptr, nullptr, 0, bridge_b, c_ws, h_bf, kH, 1);

  // --- sequential greedy decode ---
  for (int t = 0; t < kT; ++t) {
    embed_concat_kernel<<<blk((size_t)kB * kX), 256, 0, stream>>>(
        emb_table, to_concat, idx, x_bf);

    // gates[32,4096] = x @ W_ih^T + h @ W_hh^T + (b_ih + b_hh)
    wmma_gemm_kernel<<<(4 * kH) / 64, 128, lds_gates, stream>>>(
        x_bf, Wih_bf, kX, h_bf, Whh_bf, kH, bias, gates, nullptr, 4 * kH, 0);

    lstm_cell_kernel<<<blk((size_t)kB * kH), 256, 0, stream>>>(
        gates, c_ws, h_bf, out + (size_t)t * kB * kH);

    // logits straight into the projected slice of d_out, then in-place logp
    float* proj_t = out + OUT_PROJ + (size_t)t * kB * kV;
    wmma_gemm_kernel<<<kV / 64, 128, lds_logits, stream>>>(
        h_bf, outW_bf, kH, nullptr, nullptr, 0, out_b, proj_t, nullptr, kV, 0);

    logsoftmax_argmax_kernel<<<kB, 256, 0, stream>>>(proj_t, idx);
  }

  // hT = outputs[T-1]; cT = final cell state
  hipMemcpyAsync(out + OUT_HT, out + (size_t)(kT - 1) * kB * kH,
                 (size_t)kB * kH * sizeof(float), hipMemcpyDeviceToDevice, stream);
  hipMemcpyAsync(out + OUT_CT, c_ws,
                 (size_t)kB * kH * sizeof(float), hipMemcpyDeviceToDevice, stream);
}
